// GraphSAGE_34256659153681
// MI455X (gfx1250) — matrix operand order repaired and checked by reading, not compiled, not run on hardware
//
#include <hip/hip_runtime.h>
#include <hip/hip_bf16.h>

// GraphSAGE on MI455X (gfx1250, wave32).
// Memory-bound: ~1.3GB gather/scatter traffic (~56us @ 23.3TB/s HBM); the
// 12.8MB aggregation buffer lives in the 192MB L2 so float atomics resolve
// at L2. Dense node updates (~1 GFLOP total) use v_wmma_f32_16x16x32_bf16,
// fusing agg@Wl + x@Wr as one K=128 GEMM. Weights are staged in LDS
// PRE-SWIZZLED into the B-fragment lane layout so each fragment load is a
// single contiguous 32B ds read (2x ds_load_b128), and A fragments are
// built from 4x global_load_b128 per K-chunk.

typedef __attribute__((ext_vector_type(16))) __bf16 bf16x16;
typedef __attribute__((ext_vector_type(8)))  float  f32x8;

#define D_IN 64

// ---------------- degree count: cnt[dst] += 1 ----------------
__global__ void deg_kernel(const int* __restrict__ dst,
                           float* __restrict__ cnt, int E) {
    int e = blockIdx.x * blockDim.x + threadIdx.x;
    if (e < E) unsafeAtomicAdd(&cnt[dst[e]], 1.0f);
}

// ------------- scatter-add: s[dst,:] += feat[src,:] ----------
// one wave32 per edge (8 edges per 256-thread block); edge id is forced
// wave-uniform so src/dst index loads become scalar (s_load) loads.
__global__ __launch_bounds__(256)
void scatter_kernel(const float* __restrict__ feat,
                    const int* __restrict__ src,
                    const int* __restrict__ dst,
                    float* __restrict__ s, int E) {
    int e = __builtin_amdgcn_readfirstlane(blockIdx.x * 8 + (threadIdx.x >> 5));
    int lane = threadIdx.x & 31;
    if (e >= E) return;
    int sn = src[e], dn = dst[e];         // wave-uniform -> scalar loads
    int f = lane * 2;
    const float2 v = *(const float2*)(feat + (size_t)sn * D_IN + f);
    float* p = s + (size_t)dn * D_IN + f;
    unsafeAtomicAdd(p,     v.x);          // global_atomic_add_f32, hits L2
    unsafeAtomicAdd(p + 1, v.y);
}

// ---- fused SAGE node update: out = act([s/max(cnt,1) | x] @ [Wl;Wr] + b) ----
// 256 threads = 8 waves; each wave computes a 16-row x FOUT tile with
// v_wmma_f32_16x16x32_bf16, K-loop = 4 chunks of 32 over the concatenated
// K=128 (first 64 = mean-aggregated neighbors, second 64 = self features).
template<int FOUT, bool RELU>
__global__ __launch_bounds__(256)
void sage_wmma_kernel(const float* __restrict__ aggsum,
                      const float* __restrict__ cnt,
                      const float* __restrict__ xin,
                      const float* __restrict__ Wl,    // [64][FOUT]
                      const float* __restrict__ Wr,    // [64][FOUT]
                      const float* __restrict__ bias,  // [FOUT]
                      float* __restrict__ out,         // [N][FOUT]
                      int N) {
    constexpr int CT   = FOUT / 16;              // column tiles
    constexpr int NFRG = 4 * CT;                 // kt x ct fragments
    // B fragments pre-swizzled: [frag][lane][16 bf16] -> 32B contiguous/lane
    __shared__ alignas(32) __bf16 Wfrag[NFRG * 32 * 16];
    __shared__ float Bsh[FOUT];

    // Cooperative stage + swizzle of [Wl;Wr] (K=128 x FOUT) into fragment
    // layout. B order used here (same as A): lane -> col N = lane&15, element e ->
    // K_local = (lane>>4)*16 + e within the 32-deep K chunk kt.
    for (int idx = threadIdx.x; idx < NFRG * 512; idx += 256) {
        int e2   = idx & 15;
        int lane = (idx >> 4) & 31;
        int ctk  = idx >> 9;                     // kt*CT + ct
        int ct   = ctk % CT;
        int kt   = ctk / CT;
        int kg   = kt * 32 + (lane >> 4) * 16 + e2;   // 0..127
        int fout = ct * 16 + (lane & 15);
        float w = (kg < 64) ? Wl[kg * FOUT + fout]
                            : Wr[(kg - 64) * FOUT + fout];
        Wfrag[idx] = (__bf16)w;
    }
    if (threadIdx.x < FOUT) Bsh[threadIdx.x] = bias[threadIdx.x];
    __syncthreads();

    const int wave = threadIdx.x >> 5;
    const int lane = threadIdx.x & 31;
    const int half = lane >> 4;                  // lane group (K split)
    const int lm   = lane & 15;
    const int row0 = blockIdx.x * 128 + wave * 16;

    int rowA  = row0 + lm;                       // A: lane holds row M=lm
    int rowAc = rowA < N ? rowA : N - 1;         // clamp reads; stores guarded
    float inv = 1.0f / fmaxf(cnt[rowAc], 1.0f);

    const float* aRow = aggsum + (size_t)rowAc * 64;
    const float* xRow = xin    + (size_t)rowAc * 64;

    f32x8 acc[CT];
#pragma unroll
    for (int ct = 0; ct < CT; ++ct) acc[ct] = (f32x8){0,0,0,0,0,0,0,0};

#pragma unroll
    for (int kt = 0; kt < 4; ++kt) {             // K = kt*32 .. kt*32+31
        const float* base  = (kt < 2) ? aRow : xRow;
        const float  scale = (kt < 2) ? inv : 1.0f;
        const int    kbase = (kt & 1) * 32;

        // A fragment (16x32 bf16): per-lane data is one contiguous 16-float
        // run: K = kbase+half*16+[0,16) (matches B fragment layout).
        float f[16];
        *(float4*)(f + 0)  = *(const float4*)(base + kbase + half * 16);
        *(float4*)(f + 4)  = *(const float4*)(base + kbase + half * 16 + 4);
        *(float4*)(f + 8)  = *(const float4*)(base + kbase + half * 16 + 8);
        *(float4*)(f + 12) = *(const float4*)(base + kbase + half * 16 + 12);

        bf16x16 a;
#pragma unroll
        for (int i = 0; i < 16; ++i) a[i] = (__bf16)(f[i] * scale);

#pragma unroll
        for (int ct = 0; ct < CT; ++ct) {
            const bf16x16 b =
                *(const bf16x16*)&Wfrag[((kt * CT + ct) * 32 + lane) * 16];
            acc[ct] = __builtin_amdgcn_wmma_f32_16x16x32_bf16(
                false, a, false, b, (short)0, acc[ct], false, false);
        }
    }

    // C/D layout: VGPR r -> row M = half*8 + r, col = lm
#pragma unroll
    for (int ct = 0; ct < CT; ++ct) {
#pragma unroll
        for (int r = 0; r < 8; ++r) {
            int row = row0 + half * 8 + r;
            if (row < N) {
                int col = ct * 16 + lm;
                float val = acc[ct][r] + Bsh[col];
                if (RELU) val = fmaxf(val, 0.0f);
                out[(size_t)row * FOUT + col] = val;
            }
        }
    }
}

// ---------------- final FC: out = h2 @ Wfc + bfc ----------------
__global__ void fc_kernel(const float* __restrict__ h2,   // [N][16]
                          const float* __restrict__ Wfc,  // [16][2]
                          const float* __restrict__ bfc,  // [2]
                          float* __restrict__ out, int N) {
    int n = blockIdx.x * blockDim.x + threadIdx.x;
    if (n >= N) return;
    const float* h = h2 + (size_t)n * 16;
    float o0 = bfc[0], o1 = bfc[1];
#pragma unroll
    for (int k = 0; k < 16; ++k) {
        float hv = h[k];
        o0 += hv * Wfc[k * 2 + 0];
        o1 += hv * Wfc[k * 2 + 1];
    }
    out[n * 2 + 0] = o0;
    out[n * 2 + 1] = o1;
}

extern "C" void kernel_launch(void* const* d_in, const int* in_sizes, int n_in,
                              void* d_out, int out_size, void* d_ws, size_t ws_size,
                              hipStream_t stream) {
    const float* x   = (const float*)d_in[0];
    const int*   e   = (const int*)  d_in[1];
    const float* W1l = (const float*)d_in[2];
    const float* b1  = (const float*)d_in[3];
    const float* W1r = (const float*)d_in[4];
    const float* W2l = (const float*)d_in[5];
    const float* b2  = (const float*)d_in[6];
    const float* W2r = (const float*)d_in[7];
    const float* Wfc = (const float*)d_in[8];
    const float* bfc = (const float*)d_in[9];
    float* out = (float*)d_out;

    const int N = in_sizes[0] / D_IN;
    const int E = in_sizes[1] / 2;
    const int* src = e;
    const int* dst = e + E;

    // workspace: cnt[N] | s[N*64] (reused both layers) | h1[N*64] | h2[N*16]
    float* ws  = (float*)d_ws;
    float* cnt = ws;
    float* s   = cnt + N;
    float* h1  = s  + (size_t)N * 64;
    float* h2  = h1 + (size_t)N * 64;

    const int nbNodes   = (N + 127) / 128;
    const int nbEdges   = (E + 255) / 256;
    const int nbScatter = (E + 7) / 8;    // 8 edges (waves) per block

    // zero cnt + accumulator (deterministic each call; capture-safe)
    hipMemsetAsync(cnt, 0, sizeof(float) * ((size_t)N * 65), stream);

    deg_kernel<<<nbEdges, 256, 0, stream>>>(dst, cnt, E);
    scatter_kernel<<<nbScatter, 256, 0, stream>>>(x, src, dst, s, E);
    sage_wmma_kernel<64, true><<<nbNodes, 256, 0, stream>>>(
        s, cnt, x, W1l, W1r, b1, h1, N);

    hipMemsetAsync(s, 0, sizeof(float) * ((size_t)N * 64), stream);
    scatter_kernel<<<nbScatter, 256, 0, stream>>>(h1, src, dst, s, E);
    sage_wmma_kernel<16, false><<<nbNodes, 256, 0, stream>>>(
        s, cnt, h1, W2l, W2r, b2, h2, N);

    fc_kernel<<<(N + 255) / 256, 256, 0, stream>>>(h2, Wfc, bfc, out, N);
}
